// Dictionary_89172111000274
// MI455X (gfx1250) — compile-verified, matrix operand order checked by reading, not run on hardware
//
#include <hip/hip_runtime.h>
#include <math.h>

#define NUM_TASKS 1024
#define N_FEATURES 256
#define HIDDEN 64
#define BATCH 8192

typedef __attribute__((ext_vector_type(16))) __bf16 v16bf;
typedef __attribute__((ext_vector_type(8)))  __bf16 v8bf;   // one 16-byte LDS chunk
typedef __attribute__((ext_vector_type(8)))  float  v8f;

// fp32 -> bf16 round-to-nearest-even
static __device__ __forceinline__ __bf16 f2bf(float f) {
    union { float f; unsigned u; } v; v.f = f;
    unsigned r = v.u + 0x7FFFu + ((v.u >> 16) & 1u);
    unsigned short h = (unsigned short)(r >> 16);
    __bf16 o; __builtin_memcpy(&o, &h, 2); return o;
}

// Load a 16-bit WMMA fragment: two contiguous 16B chunks at K offsets
// (kk+kb) and (kk+kb+16), per ISA 16-bit A-frag layout (B identical with n as row).
static __device__ __forceinline__ v16bf load_frag(const __bf16* rowPtr, int kofs) {
    v8bf lo = *(const v8bf*)(rowPtr + kofs);
    v8bf hi = *(const v8bf*)(rowPtr + kofs + 16);
    return __builtin_shufflevector(lo, hi, 0,1,2,3,4,5,6,7,8,9,10,11,12,13,14,15);
}

// ---------------- bucketing (counting sort of sample indices by task) ----------------

__global__ void zero_counts_kernel(int* counts) {
    int i = blockIdx.x * blockDim.x + threadIdx.x;
    if (i < NUM_TASKS) counts[i] = 0;
}

__global__ void count_tasks_kernel(const int* __restrict__ task_ids, int* __restrict__ counts) {
    int i = blockIdx.x * blockDim.x + threadIdx.x;
    if (i < BATCH) atomicAdd(&counts[task_ids[i]], 1);
}

__global__ void scan_offsets_kernel(const int* __restrict__ counts,
                                    int* __restrict__ offsets, int* __restrict__ cursors) {
    __shared__ int buf[NUM_TASKS];
    int t = threadIdx.x;
    int my = counts[t];
    buf[t] = my;
    __syncthreads();
    for (int d = 1; d < NUM_TASKS; d <<= 1) {
        int v = (t >= d) ? buf[t - d] : 0;
        __syncthreads();
        buf[t] += v;
        __syncthreads();
    }
    int excl = buf[t] - my;
    offsets[t] = excl;
    cursors[t] = excl;
    if (t == NUM_TASKS - 1) offsets[NUM_TASKS] = buf[t];
}

__global__ void scatter_tasks_kernel(const int* __restrict__ task_ids,
                                     int* __restrict__ cursors, int* __restrict__ sorted) {
    int i = blockIdx.x * blockDim.x + threadIdx.x;
    if (i < BATCH) {
        int pos = atomicAdd(&cursors[task_ids[i]], 1);
        sorted[pos] = i;
    }
}

// ---------------- main per-task MLP kernel ----------------

#define XS  (N_FEATURES + 8)  // 264 elems = 528B rows (16B aligned, conflict-free)
#define WS1 (N_FEATURES + 8)  // W1^T rows: [n][k], k=0..255
#define WS2 (HIDDEN + 8)      // 72 elems = 144B rows
#define H1S (HIDDEN + 8)      // 72
#define H2S (HIDDEN + 4)      // 68 (fp32)

__global__ __launch_bounds__(128) void taskmlp_kernel(
    const float* __restrict__ x,
    const float* __restrict__ l1,
    const float* __restrict__ l2,
    const float* __restrict__ l3,
    const int*   __restrict__ offsets,
    const int*   __restrict__ sorted,
    float*       __restrict__ out)
{
    __shared__ __bf16 sW1t[HIDDEN * WS1];   // W1 transposed: [n][k]  (33 KB)
    __shared__ __bf16 sW2t[HIDDEN * WS2];   // W2 transposed: [n][k]  (9 KB)
    __shared__ float  sL3[HIDDEN];
    __shared__ __bf16 sX [16 * XS];         // [m][k]
    __shared__ __bf16 sH1[16 * H1S];        // [m][k]
    __shared__ float  sH2[16 * H2S];        // [m][k]

    const int task = blockIdx.x;
    const int off0 = offsets[task];
    const int off1 = offsets[task + 1];
    if (off0 == off1) return;   // uniform across block

    const int tidx  = threadIdx.x;
    const int wave  = tidx >> 5;
    const int lane  = tidx & 31;
    const int mr    = lane & 15;
    const int kb    = (lane >> 4) * 8;   // per-lane K base (ISA A/B frag layout)
    const int mofs  = (lane >> 4) * 8;   // C/D frag M offset for upper half-wave
    const int ntile = wave * 16;         // each wave owns one 16-col slice of N=64

    // ---- Stage weights into LDS, transposed to [n][k], fp32 -> bf16 ----
    const float* w1 = l1 + (size_t)task * (N_FEATURES * HIDDEN);
    for (int i4 = tidx * 4; i4 < N_FEATURES * HIDDEN; i4 += 128 * 4) {
        float4 w = *(const float4*)(w1 + i4);          // coalesced b128 read
        int k = i4 >> 6, n = i4 & 63;                  // emb layout: [k][n], n fastest
        sW1t[(n + 0) * WS1 + k] = f2bf(w.x);
        sW1t[(n + 1) * WS1 + k] = f2bf(w.y);
        sW1t[(n + 2) * WS1 + k] = f2bf(w.z);
        sW1t[(n + 3) * WS1 + k] = f2bf(w.w);
    }
    const float* w2 = l2 + (size_t)task * (HIDDEN * HIDDEN);
    for (int i4 = tidx * 4; i4 < HIDDEN * HIDDEN; i4 += 128 * 4) {
        float4 w = *(const float4*)(w2 + i4);
        int k = i4 >> 6, n = i4 & 63;
        sW2t[(n + 0) * WS2 + k] = f2bf(w.x);
        sW2t[(n + 1) * WS2 + k] = f2bf(w.y);
        sW2t[(n + 2) * WS2 + k] = f2bf(w.z);
        sW2t[(n + 3) * WS2 + k] = f2bf(w.w);
    }
    const float* w3 = l3 + (size_t)task * HIDDEN;
    for (int i = tidx; i < HIDDEN; i += 128) sL3[i] = w3[i];

    for (int base = off0; base < off1; base += 16) {
        const int cnt = min(16, off1 - base);

        // Stage 16 x-rows (zero-padded) as bf16
        for (int i = tidx; i < 16 * N_FEATURES; i += 128) {
            int m = i >> 8;          // N_FEATURES == 256
            int k = i & 255;
            float v = (m < cnt) ? x[(size_t)sorted[base + m] * N_FEATURES + k] : 0.0f;
            sX[m * XS + k] = f2bf(v);
        }
        __syncthreads();

        // Prefetch next tile's x rows into cache while we compute (gfx1250 global_prefetch)
        if (base + 16 < off1 && tidx < 16 && (base + 16 + tidx) < off1)
            __builtin_prefetch(&x[(size_t)sorted[base + 16 + tidx] * N_FEATURES], 0, 1);

        // ---- Layer 1: [16x256] @ [256x64]; wave -> columns [ntile, ntile+16) ----
        const __bf16* aRow1 = &sX[mr * XS];
        const __bf16* bRow1 = &sW1t[(ntile + mr) * WS1];
        v8f acc = {};
#pragma unroll
        for (int kk = 0; kk < N_FEATURES; kk += 32) {
            v16bf a = load_frag(aRow1, kk + kb);
            v16bf b = load_frag(bRow1, kk + kb);
            acc = __builtin_amdgcn_wmma_f32_16x16x32_bf16(
                      false, a, false, b, (short)0, acc, false, false);
        }
        // exact-erf GELU, store h1 as bf16 ([m][k] for layer 2)
#pragma unroll
        for (int r = 0; r < 8; ++r) {
            float v = acc[r];
            float g = 0.5f * v * (1.0f + erff(v * 0.70710678118654752f));
            sH1[(r + mofs) * H1S + ntile + mr] = f2bf(g);
        }
        __syncthreads();

        // ---- Layer 2: [16x64] @ [64x64] ----
        const __bf16* aRow2 = &sH1[mr * H1S];
        const __bf16* bRow2 = &sW2t[(ntile + mr) * WS2];
        v8f acc2 = {};
#pragma unroll
        for (int kk = 0; kk < HIDDEN; kk += 32) {
            v16bf a = load_frag(aRow2, kk + kb);
            v16bf b = load_frag(bRow2, kk + kb);
            acc2 = __builtin_amdgcn_wmma_f32_16x16x32_bf16(
                       false, a, false, b, (short)0, acc2, false, false);
        }
        // GELU, store h2 as fp32
#pragma unroll
        for (int r = 0; r < 8; ++r) {
            float v = acc2[r];
            float g = 0.5f * v * (1.0f + erff(v * 0.70710678118654752f));
            sH2[(r + mofs) * H2S + ntile + mr] = g;
        }
        __syncthreads();

        // ---- Final dot with l3 + sigmoid ----
        if (tidx < cnt) {
            float s = 0.0f;
#pragma unroll 8
            for (int k = 0; k < HIDDEN; ++k) s += sH2[tidx * H2S + k] * sL3[k];
            out[sorted[base + tidx]] = 1.0f / (1.0f + __expf(-s));
        }
        __syncthreads();  // protect sX/sH1/sH2 reuse next iteration
    }
}

// ---------------- launch ----------------

extern "C" void kernel_launch(void* const* d_in, const int* in_sizes, int n_in,
                              void* d_out, int out_size, void* d_ws, size_t ws_size,
                              hipStream_t stream) {
    (void)in_sizes; (void)n_in; (void)out_size; (void)ws_size;

    const float* x        = (const float*)d_in[0];
    const int*   task_ids = (const int*)  d_in[1];
    const float* l1       = (const float*)d_in[2];
    const float* l2       = (const float*)d_in[3];
    const float* l3       = (const float*)d_in[4];
    float*       out      = (float*)d_out;

    // workspace layout (ints): counts[1024] | offsets[1025] | cursors[1024] | sorted[8192]
    int* wsi     = (int*)d_ws;
    int* counts  = wsi;
    int* offsets = counts + NUM_TASKS;
    int* cursors = offsets + (NUM_TASKS + 1);
    int* sorted  = cursors + NUM_TASKS;

    zero_counts_kernel  <<<(NUM_TASKS + 255) / 256, 256, 0, stream>>>(counts);
    count_tasks_kernel  <<<(BATCH + 255) / 256,     256, 0, stream>>>(task_ids, counts);
    scan_offsets_kernel <<<1, NUM_TASKS,                 0, stream>>>(counts, offsets, cursors);
    scatter_tasks_kernel<<<(BATCH + 255) / 256,     256, 0, stream>>>(task_ids, cursors, sorted);
    taskmlp_kernel      <<<NUM_TASKS, 128,               0, stream>>>(x, l1, l2, l3, offsets, sorted, out);
}
